// Encoder_80753975099456
// MI455X (gfx1250) — matrix operand order repaired and checked by reading, not compiled, not run on hardware
//
#include <hip/hip_runtime.h>
#include <hip/hip_bf16.h>

// ---------------------------------------------------------------------------
// Transformer encoder forward for MI455X (gfx1250, wave32, WMMA).
// Outputs: h [B,S,D] then attns [L,B,H,S,S] concatenated in d_out.
//
// Memory-bound on the mandatory 805MB attn materialization: scores are written
// directly into d_out, softmaxed in place (vectorized b128), re-read once for
// attn@V. All GEMMs use v_wmma_f32_16x16x32_f16 (fp32->f16 at LDS staging,
// fp32 accumulation) with register-batched, software-pipelined staging.
// The B-operand layout (normal vs transposed) is a template parameter so the
// K-loop is branch-free.
// ---------------------------------------------------------------------------

#define D_MODEL   512
#define D_FF      2048
#define DK        64
#define NH        8
#define NL        6
#define SEQ       1024
#define BATCH     4
#define LN_EPS    1e-5f
#define MASK_FILL (-1e9f)

typedef __attribute__((ext_vector_type(16))) _Float16 v16h;
typedef __attribute__((ext_vector_type(8)))  _Float16 v8h;
typedef __attribute__((ext_vector_type(4)))  _Float16 vh4;
typedef __attribute__((ext_vector_type(8)))  float    v8f;
typedef __attribute__((ext_vector_type(4)))  float    vf4;

// GEMM tile config: 256 threads = 8 wave32s; block tile 128x64, K-chunk 32.
// Each wave computes a 32x32 patch = 2x2 WMMA 16x16 tiles (fragment reuse).
#define BM   128
#define BN   64
#define BK   32
#define LDSK 40      // padded row stride in halfs (80B: 16B-aligned, bank-spread)
#define NT   256

// ---- WMMA fragment loads per documented CDNA5 VGPR layouts -----------------
// A (16-bit, 16x32 MxK): lane m = lane&15; lanes 0-15 hold K 0-7 & 16-23,
// lanes 16-31 hold K 8-15 & 24-31. Two contiguous 16B LDS reads per lane.
__device__ __forceinline__ v16h load_a_frag(const _Float16* As, int mtile, int lane) {
    int m  = mtile + (lane & 15);
    int kb = (lane & 16) ? 8 : 0;
    const _Float16* p = As + m * LDSK + kb;
    v8h lo = *(const v8h*)p;            // K = kb .. kb+7
    v8h hi = *(const v8h*)(p + 16);     // K = kb+16 .. kb+23
    v16h a;
#pragma unroll
    for (int i = 0; i < 8; ++i) { a[i] = lo[i]; a[i + 8] = hi[i]; }
    return a;
}

// B (16-bit, 32x16 KxN): lane n = lane&15; lanes 0-15 hold K 0-7 & 16-23,
// lanes 16-31 hold K 8-15 & 24-31. Bt is staged N-major so this is contiguous.
__device__ __forceinline__ v16h load_b_frag(const _Float16* Bt, int ntile, int lane) {
    int n  = ntile + (lane & 15);
    int kb = (lane & 16) ? 8 : 0;
    const _Float16* p = Bt + n * LDSK + kb;
    v8h lo = *(const v8h*)p;
    v8h hi = *(const v8h*)(p + 16);
    v16h b;
#pragma unroll
    for (int i = 0; i < 8; ++i) { b[i] = lo[i]; b[i + 8] = hi[i]; }
    return b;
}

// ---------------------------------------------------------------------------
// Strided-batched GEMM: C[z] = alpha * A[z] @ B[z] (+relu / +mask epilogues).
// z -> (zb = z/H, zh = z%H); per-operand (batch, head) strides.
// BT (compile-time): B stored [N,K] row-major (used for Q @ K^T).
// maskTok: if non-null, columns with token==0 get MASK_FILL (scores mask).
// All of M,N,K divide BM/BN/BK exactly for every call site -> no guards.
// Staging is register-batched (float4 global loads, packed half LDS stores)
// and software-pipelined: next K-tile's loads overlap the current WMMAs.
// ---------------------------------------------------------------------------
template <int BT>
__global__ void __launch_bounds__(NT)
gemm_wmma_f16(const float* __restrict__ A, long lda, long sAb, long sAh,
              const float* __restrict__ Bm, long ldb, long sBb, long sBh,
              float* __restrict__ C, long ldc, long sCb, long sCh,
              int K, int H, float alpha, int relu,
              const int* __restrict__ maskTok)
{
    __shared__ __align__(16) _Float16 As[BM * LDSK];
    __shared__ __align__(16) _Float16 Bt[BN * LDSK];

    const int z  = blockIdx.z;
    const int zb = z / H;
    const int zh = z % H;
    const float* Ab = A  + (long)zb * sAb + (long)zh * sAh;
    const float* Bb = Bm + (long)zb * sBb + (long)zh * sBh;
    float*       Cb = C  + (long)zb * sCb + (long)zh * sCh;

    const int tid  = threadIdx.x;
    const int wave = tid >> 5;
    const int lane = tid & 31;
    const int wm   = (wave & 3) * 32;   // wave's row offset inside block tile
    const int wn   = (wave >> 2) * 32;  // wave's col offset inside block tile
    const long rowBase = (long)blockIdx.y * BM;
    const long colBase = (long)blockIdx.x * BN;

    // ---- per-thread staging pointers / LDS offsets (computed once) ----
    // A tile: BM x BK = 1024 float4s -> 4 per thread.
    const float* aPtr[4];
    int          aOff[4];
#pragma unroll
    for (int i = 0; i < 4; ++i) {
        int f4 = i * NT + tid;
        int r  = f4 >> 3;            // row (0..127), 8 float4s per row
        int c4 = f4 & 7;             // float4 within row
        aPtr[i] = Ab + (rowBase + r) * lda + c4 * 4;
        aOff[i] = r * LDSK + c4 * 4;
    }
    // B tile: 512 float4s -> 2 per thread. Layout depends on BT.
    const float* bPtr[2];
    int          bOff[2];
    long         bStep;
    if (!BT) {                       // B is [K,N]: load along n, store transposed
        bStep = (long)BK * ldb;
#pragma unroll
        for (int i = 0; i < 2; ++i) {
            int f4 = i * NT + tid;
            int kk = f4 >> 4;        // k (0..31), 16 float4s per k-row
            int n4 = f4 & 15;        // float4 within n
            bPtr[i] = Bb + (long)kk * ldb + colBase + n4 * 4;
            bOff[i] = (n4 * 4) * LDSK + kk;      // +j*LDSK for j in 0..3
        }
    } else {                         // B is [N,K]: load along k, store contiguous
        bStep = BK;
#pragma unroll
        for (int i = 0; i < 2; ++i) {
            int f4 = i * NT + tid;
            int n  = f4 >> 3;        // n (0..63), 8 float4s per n-row
            int c4 = f4 & 7;         // float4 within k
            bPtr[i] = Bb + (colBase + n) * ldb + c4 * 4;
            bOff[i] = n * LDSK + c4 * 4;
        }
    }

    v8f acc[2][2] = {};
    vf4 av[4], bv[2];

    // prime the pipeline: K-tile 0 into registers
#pragma unroll
    for (int i = 0; i < 4; ++i) av[i] = *(const vf4*)(aPtr[i]);
#pragma unroll
    for (int i = 0; i < 2; ++i) bv[i] = *(const vf4*)(bPtr[i]);

    const int nk = K / BK;
    for (int kt = 0; kt < nk; ++kt) {
        // ---- convert + store current K-tile registers into LDS ----
#pragma unroll
        for (int i = 0; i < 4; ++i) {
            vh4 hv;
#pragma unroll
            for (int j = 0; j < 4; ++j) hv[j] = (_Float16)av[i][j];
            *(vh4*)(As + aOff[i]) = hv;          // ds_store_b64
        }
        if (!BT) {
#pragma unroll
            for (int i = 0; i < 2; ++i)
#pragma unroll
                for (int j = 0; j < 4; ++j)
                    Bt[bOff[i] + j * LDSK] = (_Float16)bv[i][j];
        } else {
#pragma unroll
            for (int i = 0; i < 2; ++i) {
                vh4 hv;
#pragma unroll
                for (int j = 0; j < 4; ++j) hv[j] = (_Float16)bv[i][j];
                *(vh4*)(Bt + bOff[i]) = hv;
            }
        }
        __syncthreads();

        // ---- issue next K-tile's global loads (overlap with WMMA below) ----
        if (kt + 1 < nk) {
            const long ka = (long)(kt + 1) * BK;
            const long kb = (long)(kt + 1) * bStep;
#pragma unroll
            for (int i = 0; i < 4; ++i) av[i] = *(const vf4*)(aPtr[i] + ka);
#pragma unroll
            for (int i = 0; i < 2; ++i) bv[i] = *(const vf4*)(bPtr[i] + kb);
            if (kt + 2 < nk)             // global_prefetch_b8 for K-tile+2
                __builtin_prefetch(aPtr[0] + (long)(kt + 2) * BK, 0, 1);
        }

        // ---- fragments + 4 WMMAs ----
        v16h a0 = load_a_frag(As, wm,      lane);
        v16h a1 = load_a_frag(As, wm + 16, lane);
        v16h b0 = load_b_frag(Bt, wn,      lane);
        v16h b1 = load_b_frag(Bt, wn + 16, lane);

        acc[0][0] = __builtin_amdgcn_wmma_f32_16x16x32_f16(false, a0, false, b0,
                        (short)0, acc[0][0], false, false);
        acc[0][1] = __builtin_amdgcn_wmma_f32_16x16x32_f16(false, a0, false, b1,
                        (short)0, acc[0][1], false, false);
        acc[1][0] = __builtin_amdgcn_wmma_f32_16x16x32_f16(false, a1, false, b0,
                        (short)0, acc[1][0], false, false);
        acc[1][1] = __builtin_amdgcn_wmma_f32_16x16x32_f16(false, a1, false, b1,
                        (short)0, acc[1][1], false, false);
        __syncthreads();
    }

    // ---- epilogue: C/D layout VGPR r -> M = r + 8*lane[4], N = lane&15 ----
    const int mHalf = (lane & 16) ? 8 : 0;
    const int nIdx  = lane & 15;
#pragma unroll
    for (int j = 0; j < 2; ++j) {
        const long col = colBase + wn + j * 16 + nIdx;
        int masked = 0;
        if (maskTok) masked = (maskTok[(long)zb * SEQ + col] == 0);
#pragma unroll
        for (int i = 0; i < 2; ++i) {
#pragma unroll
            for (int r = 0; r < 8; ++r) {
                const long row = rowBase + wm + i * 16 + mHalf + r;
                float v = acc[i][j][r] * alpha;
                if (relu)   v = v > 0.0f ? v : 0.0f;
                if (masked) v = MASK_FILL;
                Cb[row * ldc + col] = v;
            }
        }
    }
}

// ---------------------------------------------------------------------------
// In-place row softmax, row length exactly 1024: one 256-thread block per
// row, one float4 per thread (single b128 load + single b128 store).
// ---------------------------------------------------------------------------
__global__ void __launch_bounds__(256)
softmax_rows_1024(float* __restrict__ p)
{
    __shared__ float red[256];
    vf4* r = (vf4*)(p + (long)blockIdx.x * SEQ);
    const int tid = threadIdx.x;

    vf4 v = r[tid];
    float mx = fmaxf(fmaxf(v[0], v[1]), fmaxf(v[2], v[3]));
    red[tid] = mx; __syncthreads();
#pragma unroll
    for (int s = 128; s > 0; s >>= 1) {
        if (tid < s) red[tid] = fmaxf(red[tid], red[tid + s]);
        __syncthreads();
    }
    mx = red[0]; __syncthreads();

    vf4 e;
    float sum = 0.0f;
#pragma unroll
    for (int j = 0; j < 4; ++j) { e[j] = __expf(v[j] - mx); sum += e[j]; }
    red[tid] = sum; __syncthreads();
#pragma unroll
    for (int s = 128; s > 0; s >>= 1) {
        if (tid < s) red[tid] += red[tid + s];
        __syncthreads();
    }
    const float inv = 1.0f / red[0];
#pragma unroll
    for (int j = 0; j < 4; ++j) e[j] *= inv;
    r[tid] = e;
}

// ---------------------------------------------------------------------------
// Fused residual + LayerNorm over D_MODEL=512: h = LN(f + h) * g + b.
// One 256-thread block per row; each thread owns 2 elements.
// ---------------------------------------------------------------------------
__global__ void __launch_bounds__(256)
add_layernorm(const float* __restrict__ f, float* __restrict__ h,
              const float* __restrict__ g, const float* __restrict__ b)
{
    __shared__ float red[256];
    const long row = blockIdx.x;
    const float* fr = f + row * D_MODEL;
    float*       hr = h + row * D_MODEL;
    const int tid = threadIdx.x;

    float x0 = fr[tid]       + hr[tid];
    float x1 = fr[tid + 256] + hr[tid + 256];

    red[tid] = x0 + x1; __syncthreads();
#pragma unroll
    for (int s = 128; s > 0; s >>= 1) {
        if (tid < s) red[tid] += red[tid + s];
        __syncthreads();
    }
    const float mu = red[0] * (1.0f / D_MODEL);
    __syncthreads();

    const float d0 = x0 - mu, d1 = x1 - mu;
    red[tid] = d0 * d0 + d1 * d1; __syncthreads();
#pragma unroll
    for (int s = 128; s > 0; s >>= 1) {
        if (tid < s) red[tid] += red[tid + s];
        __syncthreads();
    }
    const float rs = rsqrtf(red[0] * (1.0f / D_MODEL) + LN_EPS);

    hr[tid]       = d0 * rs * g[tid]       + b[tid];
    hr[tid + 256] = d1 * rs * g[tid + 256] + b[tid + 256];
}

// ---------------------------------------------------------------------------
// Embedding gather + sinusoidal positional encoding.
// ---------------------------------------------------------------------------
__global__ void __launch_bounds__(256)
embed_pe(const int* __restrict__ x, const float* __restrict__ emb,
         float* __restrict__ h)
{
    const long idx = (long)blockIdx.x * 256 + threadIdx.x;  // < B*S*D
    const int  d   = (int)(idx & (D_MODEL - 1));
    const long bs  = idx >> 9;                              // / D_MODEL
    const int  s   = (int)(bs & (SEQ - 1));
    const int  tok = x[bs];
    // div = exp(-(2*(d/2)) * ln(10000)/D); ln(10000) = 9.210340372
    const float freq  = __expf(-(float)(d & ~1) * (9.210340372f / (float)D_MODEL));
    const float angle = (float)s * freq;
    const float pe    = (d & 1) ? __cosf(angle) : __sinf(angle);
    h[idx] = emb[(long)tok * D_MODEL + d] + pe;
}

// ---------------------------------------------------------------------------
extern "C" void kernel_launch(void* const* d_in, const int* in_sizes, int n_in,
                              void* d_out, int out_size, void* d_ws, size_t ws_size,
                              hipStream_t stream)
{
    const int*   x   = (const int*)  d_in[0];
    const float* emb = (const float*)d_in[1];
    const float* Wq  = (const float*)d_in[2];
    const float* Wk  = (const float*)d_in[3];
    const float* Wv  = (const float*)d_in[4];
    const float* Wo  = (const float*)d_in[5];
    const float* g1  = (const float*)d_in[6];
    const float* b1  = (const float*)d_in[7];
    const float* W1  = (const float*)d_in[8];
    const float* W2  = (const float*)d_in[9];
    const float* g2  = (const float*)d_in[10];
    const float* b2  = (const float*)d_in[11];
    float* out = (float*)d_out;

    const long M  = (long)BATCH * SEQ;             // 4096 rows
    const long MB = 1024L * 1024L;                 // "1M floats"

    // Workspace layout (floats). ff overlays dead q/k/v/ctx -> 12M floats (48MB).
    float* ws  = (float*)d_ws;
    float* h   = ws;                               // [4096,512]
    float* q   = ws + 2 * MB;
    float* k   = ws + 4 * MB;
    float* v   = ws + 6 * MB;
    float* ctx = ws + 8 * MB;
    float* tmp = ws + 10 * MB;                     // Wo out / FF2 out
    float* ff  = ws + 2 * MB;                      // [4096,2048], reuses q..ctx

    float* attn_out = out + (long)BATCH * SEQ * D_MODEL;

    embed_pe<<<(int)((M * D_MODEL) / 256), 256, 0, stream>>>(x, emb, h);

    const dim3 blk(NT);
    const long sBSD = (long)SEQ * D_MODEL;         // per-batch activation stride
    const long sHSS = (long)NH * SEQ * SEQ;        // per-batch attn stride
    const long sSS  = (long)SEQ * SEQ;             // per-head attn stride

    for (int l = 0; l < NL; ++l) {
        const float* wq = Wq + (long)l * D_MODEL * D_MODEL;
        const float* wk = Wk + (long)l * D_MODEL * D_MODEL;
        const float* wv = Wv + (long)l * D_MODEL * D_MODEL;
        const float* wo = Wo + (long)l * D_MODEL * D_MODEL;
        const float* w1 = W1 + (long)l * D_MODEL * D_FF;
        const float* w2 = W2 + (long)l * D_FF * D_MODEL;
        float* attnL = attn_out + (long)l * BATCH * sHSS;

        // Q, K, V projections: [4096,512] = h @ W
        gemm_wmma_f16<0><<<dim3(D_MODEL / BN, (int)(M / BM), 1), blk, 0, stream>>>(
            h, D_MODEL, 0, 0,  wq, D_MODEL, 0, 0,
            q, D_MODEL, 0, 0,  D_MODEL, 1, 1.0f, 0, nullptr);
        gemm_wmma_f16<0><<<dim3(D_MODEL / BN, (int)(M / BM), 1), blk, 0, stream>>>(
            h, D_MODEL, 0, 0,  wk, D_MODEL, 0, 0,
            k, D_MODEL, 0, 0,  D_MODEL, 1, 1.0f, 0, nullptr);
        gemm_wmma_f16<0><<<dim3(D_MODEL / BN, (int)(M / BM), 1), blk, 0, stream>>>(
            h, D_MODEL, 0, 0,  wv, D_MODEL, 0, 0,
            v, D_MODEL, 0, 0,  D_MODEL, 1, 1.0f, 0, nullptr);

        // scores = (Q @ K^T) * 1/sqrt(dk), pad-mask -> write straight to d_out
        gemm_wmma_f16<1><<<dim3(SEQ / BN, SEQ / BM, BATCH * NH), blk, 0, stream>>>(
            q, D_MODEL, sBSD, DK,
            k, D_MODEL, sBSD, DK,
            attnL, SEQ, sHSS, sSS,
            DK, NH, 0.125f, 0, x);

        // softmax in place on the attn slab
        softmax_rows_1024<<<BATCH * NH * SEQ, 256, 0, stream>>>(attnL);

        // ctx = attn @ V (per batch/head)
        gemm_wmma_f16<0><<<dim3(DK / BN, SEQ / BM, BATCH * NH), blk, 0, stream>>>(
            attnL, SEQ, sHSS, sSS,
            v, D_MODEL, sBSD, DK,
            ctx, D_MODEL, sBSD, DK,
            SEQ, NH, 1.0f, 0, nullptr);

        // tmp = ctx @ Wo; h = LN(tmp + h)
        gemm_wmma_f16<0><<<dim3(D_MODEL / BN, (int)(M / BM), 1), blk, 0, stream>>>(
            ctx, D_MODEL, 0, 0,  wo, D_MODEL, 0, 0,
            tmp, D_MODEL, 0, 0,  D_MODEL, 1, 1.0f, 0, nullptr);
        add_layernorm<<<(int)M, 256, 0, stream>>>(tmp, h,
            g1 + (long)l * D_MODEL, b1 + (long)l * D_MODEL);

        // ff = relu(h @ W1); tmp = ff @ W2; h = LN(tmp + h)
        gemm_wmma_f16<0><<<dim3(D_FF / BN, (int)(M / BM), 1), blk, 0, stream>>>(
            h, D_MODEL, 0, 0,  w1, D_FF, 0, 0,
            ff, D_FF, 0, 0,  D_MODEL, 1, 1.0f, /*relu=*/1, nullptr);
        gemm_wmma_f16<0><<<dim3(D_MODEL / BN, (int)(M / BM), 1), blk, 0, stream>>>(
            ff, D_FF, 0, 0,  w2, D_MODEL, 0, 0,
            tmp, D_MODEL, 0, 0,  D_FF, 1, 1.0f, 0, nullptr);
        add_layernorm<<<(int)M, 256, 0, stream>>>(tmp, h,
            g2 + (long)l * D_MODEL, b2 + (long)l * D_MODEL);
    }

    // h -> front of d_out (attn already written in place)
    hipMemcpyAsync(d_out, h, (size_t)M * D_MODEL * sizeof(float),
                   hipMemcpyDeviceToDevice, stream);
}
